// MASNN_25391846654708
// MI455X (gfx1250) — matrix operand order repaired and checked by reading, not compiled, not run on hardware
//
#include <hip/hip_runtime.h>
#include <hip/hip_bf16.h>

typedef __attribute__((ext_vector_type(16))) __bf16 v16bf;
typedef __attribute__((ext_vector_type(8)))  __bf16 v8bf;
typedef __attribute__((ext_vector_type(8)))  float  v8f;

#define DIMV 256
#define LN_EPS 1e-5f

__device__ __forceinline__ v8f wmma_bf16(v16bf a, v16bf b, v8f c) {
    return __builtin_amdgcn_wmma_f32_16x16x32_bf16(false, a, false, b, (short)0, c, false, false);
}

__device__ __forceinline__ void atomAddF(float* p, float v) {
    __hip_atomic_fetch_add(p, v, __ATOMIC_RELAXED, __HIP_MEMORY_SCOPE_AGENT);
}

__device__ __forceinline__ float sigmoidf_(float x) { return 1.0f / (1.0f + __expf(-x)); }

// Build a 16x32 bf16 A-fragment for this lane from a row-major [.,512] matrix.
// A layout: lanes 0-15 hold row M=lane, K 0-15;
// lanes 16-31 hold row M=lane-16, K 16-31 (contiguous, matches B).
__device__ __forceinline__ v16bf load_a_frag(const __bf16* __restrict__ rowbase,
                                             int ko, int half) {
    v8bf lo = *(const v8bf*)(rowbase + ko + half * 16);
    v8bf hi = *(const v8bf*)(rowbase + ko + half * 16 + 8);
    v16bf a;
#pragma unroll
    for (int i = 0; i < 8; ++i) { a[i] = lo[i]; a[8 + i] = hi[i]; }
    return a;
}

// Convert f32 weights to bf16 once. First ngate elems -> Wgb, rest -> Wub.
__global__ void k_cvt_weights(const float* __restrict__ Wg, const float* __restrict__ Wu,
                              __bf16* __restrict__ Wgb, __bf16* __restrict__ Wub,
                              int ngate, int total) {
    int i = blockIdx.x * 256 + threadIdx.x;
    if (i >= total) return;
    if (i < ngate) Wgb[i] = (__bf16)Wg[i];
    else           Wub[i - ngate] = (__bf16)Wu[i - ngate];
}

// Per edge: gather h[src], h[dst]; LayerNorm over the 512-wide concat; emit bf16 inp1.
// Thread 0 also bumps the destination count (masked).
__global__ void k_gather_ln1(const float* __restrict__ h, const int* __restrict__ src,
                             const int* __restrict__ dst, const unsigned char* __restrict__ mask,
                             const float* __restrict__ g, const float* __restrict__ b,
                             __bf16* __restrict__ inp1, float* __restrict__ cnt) {
    int e = blockIdx.x;
    int i = threadIdx.x;                       // 0..255
    int s = src[e], t = dst[e];
    float x  = h[(size_t)s * DIMV + i];
    float hd = h[(size_t)t * DIMV + i];

    __shared__ float s_sum[256], s_sq[256], s_mb[2];
    s_sum[i] = x + hd;
    s_sq[i]  = x * x + hd * hd;
    __syncthreads();
    for (int st = 128; st > 0; st >>= 1) {
        if (i < st) { s_sum[i] += s_sum[i + st]; s_sq[i] += s_sq[i + st]; }
        __syncthreads();
    }
    if (i == 0) {
        float m = s_sum[0] * (1.0f / 512.0f);
        float v = s_sq[0] * (1.0f / 512.0f) - m * m;
        s_mb[0] = m; s_mb[1] = rsqrtf(v + LN_EPS);
    }
    __syncthreads();
    float m = s_mb[0], rs = s_mb[1];
    size_t base = (size_t)e * 512;
    inp1[base + i]        = (__bf16)(((x  - m) * rs) * g[i]        + b[i]);
    inp1[base + 256 + i]  = (__bf16)(((hd - m) * rs) * g[256 + i]  + b[256 + i]);
    if (i == 0 && mask[e]) atomAddF(&cnt[t], 1.0f);
}

// Fused gates GEMM: one wave per (32-edge x 16-dim) tile. For each of the 5 gate
// column tiles, the B fragment is loaded once and used by two WMMAs (two 16-edge
// row tiles) -> 14 b128 loads per 10 WMMAs. K=512 -> 16 k-steps. sigmoid / 3-way
// softmax applied in-register; never materializes the [E,1280] gates tensor.
__global__ void __launch_bounds__(32)
k_gates(const __bf16* __restrict__ inp1, const __bf16* __restrict__ Wgb,
        const float* __restrict__ bg, const float* __restrict__ h,
        const int* __restrict__ src, const int* __restrict__ dst,
        __bf16* __restrict__ cat2, float* __restrict__ xz, float* __restrict__ z2b, int E) {
    int eb   = blockIdx.x * 32;
    int db   = blockIdx.y * 16;
    int lane = threadIdx.x;
    int half = lane >> 4;
    int ln   = lane & 15;
    int n    = db + ln;

    int arow0 = eb + ln;      if (arow0 > E - 1) arow0 = E - 1;
    int arow1 = eb + 16 + ln; if (arow1 > E - 1) arow1 = E - 1;
    const __bf16* Abase0 = inp1 + (size_t)arow0 * 512;
    const __bf16* Abase1 = inp1 + (size_t)arow1 * 512;

    v8f acc0[5] = {};
    v8f acc1[5] = {};
    for (int k = 0; k < 16; ++k) {
        int ko = k * 32;
        v16bf a0 = load_a_frag(Abase0, ko, half);
        v16bf a1 = load_a_frag(Abase1, ko, half);
#pragma unroll
        for (int c = 0; c < 5; ++c) {
            int col = c * 256 + n;                       // row of W_g == column of B
            v16bf bm = *(const v16bf*)(Wgb + (size_t)col * 512 + ko + half * 16);
            acc0[c] = wmma_bf16(a0, bm, acc0[c]);
            acc1[c] = wmma_bf16(a1, bm, acc1[c]);
        }
    }

    float bias[5];
#pragma unroll
    for (int c = 0; c < 5; ++c) bias[c] = bg[c * 256 + n];

#pragma unroll
    for (int t2 = 0; t2 < 2; ++t2) {
        const v8f* acc = t2 ? acc1 : acc0;
#pragma unroll
        for (int r = 0; r < 8; ++r) {
            int m = eb + t2 * 16 + half * 8 + r;         // C/D layout: lanes16-31 -> M=8..15
            if (m >= E) continue;
            int s = src[m], t = dst[m];
            float x  = h[(size_t)s * DIMV + n];
            float hd = h[(size_t)t * DIMV + n];
            float g0 = acc[0][r] + bias[0];
            float g1 = acc[1][r] + bias[1];
            float g2 = acc[2][r] + bias[2];
            float g3 = acc[3][r] + bias[3];
            float g4 = acc[4][r] + bias[4];              // GATE_BIAS == 0
            float rx = sigmoidf_(g0), rh = sigmoidf_(g1);
            float mx = fmaxf(g2, fmaxf(g3, g4));
            float e2 = __expf(g2 - mx), e3 = __expf(g3 - mx), e4 = __expf(g4 - mx);
            float inv = 1.0f / (e2 + e3 + e4);
            cat2[(size_t)m * 512 + n]       = (__bf16)(x * rx);
            cat2[(size_t)m * 512 + 256 + n] = (__bf16)(hd * rh);
            xz [(size_t)m * DIMV + n] = x * (e2 * inv) + hd * (e3 * inv);
            z2b[(size_t)m * DIMV + n] = e4 * inv;
        }
    }
}

// LayerNorm2 in place over cat2 rows (bf16 in/out, f32 stats).
__global__ void k_ln2(__bf16* __restrict__ cat2, const float* __restrict__ g,
                      const float* __restrict__ b) {
    int e = blockIdx.x;
    int i = threadIdx.x;
    size_t base = (size_t)e * 512;
    float a0 = (float)cat2[base + i];
    float a1 = (float)cat2[base + 256 + i];

    __shared__ float s_sum[256], s_sq[256], s_mb[2];
    s_sum[i] = a0 + a1;
    s_sq[i]  = a0 * a0 + a1 * a1;
    __syncthreads();
    for (int st = 128; st > 0; st >>= 1) {
        if (i < st) { s_sum[i] += s_sum[i + st]; s_sq[i] += s_sq[i + st]; }
        __syncthreads();
    }
    if (i == 0) {
        float m = s_sum[0] * (1.0f / 512.0f);
        float v = s_sq[0] * (1.0f / 512.0f) - m * m;
        s_mb[0] = m; s_mb[1] = rsqrtf(v + LN_EPS);
    }
    __syncthreads();
    float m = s_mb[0], rs = s_mb[1];
    cat2[base + i]       = (__bf16)(((a0 - m) * rs) * g[i]       + b[i]);
    cat2[base + 256 + i] = (__bf16)(((a1 - m) * rs) * g[256 + i] + b[256 + i]);
}

// Candidate GEMM + tanh + combine + masked atomic scatter into per-node sums.
// One wave per (64-edge x 16-dim) tile: B fragment loaded once per k-step and
// shared by four row-tile WMMAs -> 10 b128 loads per 4 WMMAs.
__global__ void __launch_bounds__(32)
k_cand(const __bf16* __restrict__ inp2, const __bf16* __restrict__ Wub,
       const float* __restrict__ bu, const float* __restrict__ xz,
       const float* __restrict__ z2b, const int* __restrict__ dst,
       const unsigned char* __restrict__ mask, float* __restrict__ sums, int E) {
    int eb   = blockIdx.x * 64;
    int db   = blockIdx.y * 16;
    int lane = threadIdx.x;
    int half = lane >> 4;
    int ln   = lane & 15;
    int n    = db + ln;

    const __bf16* Abase[4];
#pragma unroll
    for (int t4 = 0; t4 < 4; ++t4) {
        int arow = eb + t4 * 16 + ln;
        if (arow > E - 1) arow = E - 1;
        Abase[t4] = inp2 + (size_t)arow * 512;
    }

    v8f acc[4] = {};
    for (int k = 0; k < 16; ++k) {
        int ko = k * 32;
        v16bf bm = *(const v16bf*)(Wub + (size_t)n * 512 + ko + half * 16);
#pragma unroll
        for (int t4 = 0; t4 < 4; ++t4) {
            v16bf a = load_a_frag(Abase[t4], ko, half);
            acc[t4] = wmma_bf16(a, bm, acc[t4]);
        }
    }

    float bias = bu[n];
#pragma unroll
    for (int t4 = 0; t4 < 4; ++t4) {
#pragma unroll
        for (int r = 0; r < 8; ++r) {
            int m = eb + t4 * 16 + half * 8 + r;
            if (m >= E) continue;
            if (!mask[m]) continue;
            float u  = tanhf(acc[t4][r] + bias);
            float he = xz[(size_t)m * DIMV + n] + z2b[(size_t)m * DIMV + n] * u;
            atomAddF(&sums[(size_t)dst[m] * DIMV + n], he);
        }
    }
}

// h_cur[n,:] = sums[n,:]/cnt[n] where cnt>0, else unchanged.
__global__ void k_apply(float* __restrict__ h, const float* __restrict__ sums,
                        const float* __restrict__ cnt, int total) {
    int idx = blockIdx.x * 256 + threadIdx.x;
    if (idx >= total) return;
    float c = cnt[idx >> 8];
    if (c > 0.0f) h[idx] = sums[idx] / c;
}

extern "C" void kernel_launch(void* const* d_in, const int* in_sizes, int n_in,
                              void* d_out, int out_size, void* d_ws, size_t ws_size,
                              hipStream_t stream) {
    const float* h    = (const float*)d_in[0];
    const float* W_g  = (const float*)d_in[1];
    const float* b_g  = (const float*)d_in[2];
    const float* W_u  = (const float*)d_in[3];
    const float* b_u  = (const float*)d_in[4];
    const float* ln1g = (const float*)d_in[5];
    const float* ln1b = (const float*)d_in[6];
    const float* ln2g = (const float*)d_in[7];
    const float* ln2b = (const float*)d_in[8];
    const int*   src  = (const int*)d_in[9];
    const int*   dst  = (const int*)d_in[10];
    const unsigned char* mask = (const unsigned char*)d_in[11];

    const int D = DIMV;
    const int N = in_sizes[0] / D;       // 50000
    const int F = 8;
    const int E = in_sizes[9] / F;       // 100000

    char* p = (char*)d_ws;
    auto carve = [&](size_t bytes) -> char* {
        char* r = p;
        p += (bytes + 255) & ~((size_t)255);
        return r;
    };
    float*  h_cur = (float*)carve((size_t)N * D * 4);
    __bf16* Wgb   = (__bf16*)carve((size_t)5 * D * 2 * D * 2);
    __bf16* Wub   = (__bf16*)carve((size_t)D * 2 * D * 2);
    __bf16* inp1  = (__bf16*)carve((size_t)E * 2 * D * 2);
    __bf16* cat2  = (__bf16*)carve((size_t)E * 2 * D * 2);
    float*  xz    = (float*)carve((size_t)E * D * 4);
    float*  z2b   = (float*)carve((size_t)E * D * 4);
    float*  sums  = (float*)carve((size_t)N * D * 4);
    float*  cnt   = (float*)carve((size_t)N * 4);

    hipMemcpyAsync(h_cur, h, (size_t)N * D * 4, hipMemcpyDeviceToDevice, stream);

    {
        int ngate = 5 * D * 2 * D;           // 1280*512
        int total = ngate + D * 2 * D;       // + 256*512
        k_cvt_weights<<<(total + 255) / 256, 256, 0, stream>>>(W_g, W_u, Wgb, Wub, ngate, total);
    }

    int et32 = (E + 31) / 32;
    int et64 = (E + 63) / 64;
    for (int f = 0; f < F; ++f) {
        const int* sf = src + (size_t)f * E;
        const int* tf = dst + (size_t)f * E;
        const unsigned char* mf = mask + (size_t)f * E;

        hipMemsetAsync(sums, 0, (size_t)N * D * 4, stream);
        hipMemsetAsync(cnt, 0, (size_t)N * 4, stream);

        k_gather_ln1<<<E, 256, 0, stream>>>(h_cur, sf, tf, mf, ln1g, ln1b, inp1, cnt);
        k_gates<<<dim3(et32, 16), 32, 0, stream>>>(inp1, Wgb, b_g, h_cur, sf, tf, cat2, xz, z2b, E);
        k_ln2<<<E, 256, 0, stream>>>(cat2, ln2g, ln2b);
        k_cand<<<dim3(et64, 16), 32, 0, stream>>>(cat2, Wub, b_u, xz, z2b, tf, mf, sums, E);
        k_apply<<<((N * D) + 255) / 256, 256, 0, stream>>>(h_cur, sums, cnt, N * D);
    }

    hipMemcpyAsync(d_out, h_cur, (size_t)N * D * 4, hipMemcpyDeviceToDevice, stream);
}
